// Seq2SeqAttention_23398981829194
// MI455X (gfx1250) — matrix operand order repaired and checked by reading, not compiled, not run on hardware
//
#include <hip/hip_runtime.h>
#include <hip/hip_bf16.h>
#include <math.h>

typedef __attribute__((ext_vector_type(16))) __bf16 v16bf;
typedef __attribute__((ext_vector_type(8)))  float  v8f;
typedef __attribute__((ext_vector_type(4)))  int    v4i;

#define NEGVAL (-10000000000.0f)
#define LDS_STRIDE 36   // 16x36 f32 tile: rows 144B apart -> 16B aligned, 2-way bank conflict

// ---------- helpers ----------

static __device__ __forceinline__ float fast_tanhf(float x) {
#if __has_builtin(__builtin_amdgcn_tanhf)
  return __builtin_amdgcn_tanhf(x);           // v_tanh_f32 (gfx1250 trans pipe)
#elif __has_builtin(__builtin_amdgcn_tanh_f32)
  return __builtin_amdgcn_tanh_f32(x);
#else
  return tanhf(x);
#endif
}

static __device__ __forceinline__ v8f wmma_bf16(v16bf a, v16bf b, v8f c) {
  // D = A(16x32 bf16) * B(32x16 bf16) + C(16x16 f32)
  return __builtin_amdgcn_wmma_f32_16x16x32_bf16(false, a, false, b, (short)0, c,
                                                 false, false);
}

static __device__ __forceinline__ __bf16 bf16_hi(float x) { return (__bf16)x; }
static __device__ __forceinline__ __bf16 bf16_lo(float x, __bf16 h) {
  return (__bf16)(x - (float)h);
}

// ---------- async global->LDS staging (gfx1250 ASYNCcnt path) ----------
#if __has_builtin(__builtin_amdgcn_global_load_async_to_lds_b128)
#define HAVE_ASYNC_LDS 1
#else
#define HAVE_ASYNC_LDS 0
#endif

#if HAVE_ASYNC_LDS
typedef __attribute__((address_space(1))) v4i gas_v4i;   // global
typedef __attribute__((address_space(3))) v4i las_v4i;   // LDS

static __device__ __forceinline__ void async_ld_b128(float* lds, const float* g) {
  __builtin_amdgcn_global_load_async_to_lds_b128(
      (gas_v4i*)(uintptr_t)g,
      (las_v4i*)(unsigned)(uintptr_t)lds,
      0, 0);
}
static __device__ __forceinline__ void async_wait0() {
#if __has_builtin(__builtin_amdgcn_s_wait_asynccnt)
  __builtin_amdgcn_s_wait_asynccnt(0);
#else
  asm volatile("s_wait_asynccnt 0" ::: "memory");
#endif
}
#endif

// Stage a 16x32 f32 A-tile (row stride 512 in global) into padded LDS.
static __device__ __forceinline__ void stage_a_tile(float* lds_a,
                                                    const float* src, int kk0,
                                                    int t) {
#if HAVE_ASYNC_LDS
  if (t < 128) {                         // 128 lanes x b128 = 2KB tile
    const int r = t >> 3;                // 0..15
    const int c = (t & 7) << 2;          // 0,4,...,28
    async_ld_b128(&lds_a[r * LDS_STRIDE + c], src + (size_t)r * 512 + kk0 + c);
  }
  async_wait0();
#else
#pragma unroll
  for (int i = 0; i < 2; ++i) {
    int idx = t + i * 256;
    int r = idx >> 5, c = idx & 31;
    lds_a[r * LDS_STRIDE + c] = src[(size_t)r * 512 + kk0 + c];
  }
#endif
}

// Build A fragment (16x32, M x K) hi/lo from padded LDS tile lds[m*LDS_STRIDE + k].
// Lane L<16: M=L, elems 0..15 -> K=0..15.
// Lane L>=16: M=L-16, elems 0..15 -> K=16..31.
static __device__ __forceinline__ void build_a_frag(const float* lds, int lane,
                                                    v16bf& a_hi, v16bf& a_lo) {
  const int m   = lane & 15;
  const int kb0 = (lane < 16) ? 0 : 16;
#pragma unroll
  for (int j = 0; j < 8; ++j) {
    float x0 = lds[m * LDS_STRIDE + kb0 + j];
    float x1 = lds[m * LDS_STRIDE + 8 + kb0 + j];
    __bf16 h0 = bf16_hi(x0), h1 = bf16_hi(x1);
    a_hi[j]     = h0; a_lo[j]     = bf16_lo(x0, h0);
    a_hi[8 + j] = h1; a_lo[8 + j] = bf16_lo(x1, h1);
  }
}

// ---------- kernel 1: fused projections  q_proj / k_proj = X @ W_partᵀ ----------
// rows 0..1023   : query rows, W columns [0,512)
// rows 1024..5119: kv rows,    W columns [512,1024)
// Each block: 16 rows x 256 d_attn cols; 8 waves, each wave owns two 16x16 N-tiles.
__global__ __launch_bounds__(256) void proj_kernel(
    const float* __restrict__ query, const float* __restrict__ kv,
    const float* __restrict__ W, float* __restrict__ q_proj,
    float* __restrict__ k_proj) {
  __shared__ float lds_a[16 * LDS_STRIDE];

  const int rb = blockIdx.x;            // 0..319
  const bool is_kv = rb >= 64;
  const float* src = is_kv ? (kv + (size_t)(rb - 64) * 16 * 512)
                           : (query + (size_t)rb * 16 * 512);
  float* dst = is_kv ? (k_proj + (size_t)(rb - 64) * 16 * 256)
                     : (q_proj + (size_t)rb * 16 * 256);
  const int wcol = is_kv ? 512 : 0;

  const int t    = threadIdx.x;
  const int wave = t >> 5;
  const int lane = t & 31;

  v8f acc0 = {}, acc1 = {};

  for (int kk0 = 0; kk0 < 512; kk0 += 32) {
    __syncthreads();
    stage_a_tile(lds_a, src, kk0, t);
    __syncthreads();

    v16bf a_hi, a_lo;
    build_a_frag(lds_a, lane, a_hi, a_lo);

#pragma unroll
    for (int tt = 0; tt < 2; ++tt) {
      const int nb = wave * 2 + tt;                 // 16-wide N tile index, 0..15
      const int n  = nb * 16 + (lane & 15);
      const int koff = (lane < 16) ? 0 : 16;
      const float* wrow = W + (size_t)n * 1024 + wcol + kk0 + koff;
      v16bf b_hi, b_lo;
#pragma unroll
      for (int j = 0; j < 16; ++j) {
        float x = wrow[j];
        __bf16 h = bf16_hi(x);
        b_hi[j] = h;
        b_lo[j] = bf16_lo(x, h);
      }

      v8f& acc = tt ? acc1 : acc0;
      acc = wmma_bf16(a_lo, b_hi, acc);   // small terms first
      acc = wmma_bf16(a_hi, b_lo, acc);
      acc = wmma_bf16(a_hi, b_hi, acc);
    }
  }

  // writeback: C layout — lanes 0-15: M=r, lanes 16-31: M=8+r (VGPR r)
  const int m0   = (lane < 16) ? 0 : 8;
#pragma unroll
  for (int tt = 0; tt < 2; ++tt) {
    const v8f acc = tt ? acc1 : acc0;
    const int ncol = (wave * 2 + tt) * 16 + (lane & 15);
#pragma unroll
    for (int r = 0; r < 8; ++r)
      dst[(size_t)(m0 + r) * 256 + ncol] = acc[r];
  }
}

// ---------- kernel 2: score = v . tanh(qp + kp + b), masked softmax over K ----------
// One block (8 waves) per (b,q). Wave handles one k at a time, lanes split a=0..255.
__global__ __launch_bounds__(256) void score_softmax_kernel(
    const float* __restrict__ q_proj, const float* __restrict__ k_proj,
    const float* __restrict__ bias, const float* __restrict__ vvec,
    const int* __restrict__ query_len, const int* __restrict__ kv_len,
    float* __restrict__ attn) {
  __shared__ float qb[256];
  __shared__ float vv[256];
  __shared__ float sc[512];
  __shared__ float red[8];

  const int bb = blockIdx.x >> 7;       // B = 8
  const int q  = blockIdx.x & 127;      // Q = 128
  const int t    = threadIdx.x;
  const int wave = t >> 5;
  const int lane = t & 31;

  qb[t] = q_proj[((size_t)(bb * 128 + q)) * 256 + t] + bias[t];
  vv[t] = vvec[t];
  __syncthreads();

  const bool qvalid = q < query_len[bb];
  const int  klen   = kv_len[bb];

  for (int k = wave; k < 512; k += 8) {
    const float* kp = k_proj + ((size_t)(bb * 512 + k)) * 256;
    float s = 0.0f;
#pragma unroll
    for (int j = 0; j < 8; ++j) {
      const int a = lane + j * 32;      // coalesced across the wave
      s += vv[a] * fast_tanhf(qb[a] + kp[a]);
    }
#pragma unroll
    for (int off = 16; off > 0; off >>= 1) s += __shfl_xor(s, off, 32);
    if (lane == 0) sc[k] = (qvalid && (k < klen)) ? s : NEGVAL;
  }
  __syncthreads();

  // --- softmax over sc[0..511]; each thread owns entries t and t+256 ---
  float m = fmaxf(sc[t], sc[t + 256]);
#pragma unroll
  for (int off = 16; off > 0; off >>= 1) m = fmaxf(m, __shfl_xor(m, off, 32));
  if (lane == 0) red[wave] = m;
  __syncthreads();
  if (t == 0) {
    float mm = red[0];
#pragma unroll
    for (int i = 1; i < 8; ++i) mm = fmaxf(mm, red[i]);
    red[0] = mm;
  }
  __syncthreads();
  const float smax = red[0];
  __syncthreads();

  const float e0 = __expf(sc[t] - smax);
  const float e1 = __expf(sc[t + 256] - smax);
  sc[t] = e0;
  sc[t + 256] = e1;
  float s = e0 + e1;
#pragma unroll
  for (int off = 16; off > 0; off >>= 1) s += __shfl_xor(s, off, 32);
  if (lane == 0) red[wave] = s;
  __syncthreads();
  if (t == 0) {
    float ss = red[0];
#pragma unroll
    for (int i = 1; i < 8; ++i) ss += red[i];
    red[0] = ss;
  }
  __syncthreads();
  const float inv = 1.0f / red[0];

  float* arow = attn + ((size_t)(bb * 128 + q)) * 512;
  arow[t]       = sc[t] * inv;
  arow[t + 256] = sc[t + 256] * inv;
}

// ---------- kernel 3: out[b] = attn[b] (128x512) @ kv[b] (512x512) ----------
// Block: 16 rows x 256 cols; 8 waves x two 16x16 N-tiles; K-loop in steps of 32.
__global__ __launch_bounds__(256) void out_kernel(
    const float* __restrict__ attn, const float* __restrict__ kv,
    float* __restrict__ out) {
  __shared__ float lds_a[16 * LDS_STRIDE];

  const int blk   = blockIdx.x;         // 128 blocks
  const int b     = blk >> 4;
  const int mb    = (blk & 15) >> 1;    // 0..7 -> 16-row tile of Q=128
  const int nhalf = blk & 1;            // 0..1 -> 256-col half of d_kv=512

  const int t    = threadIdx.x;
  const int wave = t >> 5;
  const int lane = t & 31;

  const float* arows = attn + ((size_t)b * 128 + mb * 16) * 512;
  const float* kvb   = kv + (size_t)b * 512 * 512;
  float* dst = out + ((size_t)b * 128 + mb * 16) * 512;

  v8f acc0 = {}, acc1 = {};

  for (int kk0 = 0; kk0 < 512; kk0 += 32) {
    __syncthreads();
    stage_a_tile(lds_a, arows, kk0, t);
    __syncthreads();

    v16bf a_hi, a_lo;
    build_a_frag(lds_a, lane, a_hi, a_lo);

    // prefetch next kv K-tile into cache (global_prefetch_b8)
    if (kk0 + 32 < 512)
      __builtin_prefetch(kvb + (size_t)(kk0 + 32) * 512 + nhalf * 256 + lane * 8, 0, 1);

#pragma unroll
    for (int tt = 0; tt < 2; ++tt) {
      const int n    = nhalf * 256 + (wave * 2 + tt) * 16 + (lane & 15);
      const int koff = (lane < 16) ? 0 : 16;
      const float* bcol = kvb + (size_t)(kk0 + koff) * 512 + n;
      v16bf b_hi, b_lo;
#pragma unroll
      for (int j = 0; j < 16; ++j) {
        float x = bcol[(size_t)j * 512];
        __bf16 h = bf16_hi(x);
        b_hi[j] = h;
        b_lo[j] = bf16_lo(x, h);
      }

      v8f& acc = tt ? acc1 : acc0;
      acc = wmma_bf16(a_lo, b_hi, acc);
      acc = wmma_bf16(a_hi, b_lo, acc);
      acc = wmma_bf16(a_hi, b_hi, acc);
    }
  }

  const int m0 = (lane < 16) ? 0 : 8;
#pragma unroll
  for (int tt = 0; tt < 2; ++tt) {
    const v8f acc = tt ? acc1 : acc0;
    const int ncol = nhalf * 256 + (wave * 2 + tt) * 16 + (lane & 15);
#pragma unroll
    for (int r = 0; r < 8; ++r)
      dst[(size_t)(m0 + r) * 512 + ncol] = acc[r];
  }
}

// ---------- launch ----------
extern "C" void kernel_launch(void* const* d_in, const int* in_sizes, int n_in,
                              void* d_out, int out_size, void* d_ws, size_t ws_size,
                              hipStream_t stream) {
  const float* query = (const float*)d_in[0];   // (8,128,512)
  const float* kv    = (const float*)d_in[1];   // (8,512,512)
  const float* W     = (const float*)d_in[2];   // (256,1024)
  const float* bias  = (const float*)d_in[3];   // (256,)
  const float* vvec  = (const float*)d_in[4];   // (256,)
  const int*   qlen  = (const int*)d_in[5];     // (8,)
  const int*   klen  = (const int*)d_in[6];     // (8,)
  float* out = (float*)d_out;                   // (8,128,512) f32

  float* q_proj = (float*)d_ws;                 // 8*128*256 f32 = 1 MB
  float* k_proj = q_proj + 8 * 128 * 256;       // 8*512*256 f32 = 4 MB
  float* attn   = k_proj + 8 * 512 * 256;       // 8*128*512 f32 = 2 MB

  proj_kernel<<<320, 256, 0, stream>>>(query, kv, W, q_proj, k_proj);
  score_softmax_kernel<<<1024, 256, 0, stream>>>(q_proj, k_proj, bias, vvec,
                                                 qlen, klen, attn);
  out_kernel<<<128, 256, 0, stream>>>(attn, kv, out);
}
